// VectorQuantizedVAE_53884659696042
// MI455X (gfx1250) — matrix operand order repaired and checked by reading, not compiled, not run on hardware
//
#include <hip/hip_runtime.h>
#include <hip/hip_bf16.h>
#include <cstdint>

// ---------------------------------------------------------------------------
// VQ-VAE forward for MI455X (gfx1250, wave32, WMMA bf16 16x16x32, f32 accum)
//  - weights pre-converted once to bf16 [N][K] (fragment-ready, transposed)
//  - GEMM k-loop double-buffered: async B copy (t+1) + A loads (t+1)
//    overlapped with WMMA on tile t; one barrier per iteration
//  - B tiles staged with global_load_async_to_lds_b128 (ASYNCcnt)
//  - A tiles f32->bf16 via v_perm_b32 packing, ds_store_b64
//  - fragments loaded as ds_load_b128 pairs per ISA 7.12.2 layouts
// ---------------------------------------------------------------------------

typedef __bf16 bf16_t;
typedef bf16_t   v16bf __attribute__((ext_vector_type(16)));
typedef float    v8f   __attribute__((ext_vector_type(8)));
typedef uint32_t u32;
typedef u32      v2u   __attribute__((ext_vector_type(2)));
typedef u32      v4u   __attribute__((ext_vector_type(4)));

#define B_SZ   4096
#define IN_SZ  8192
#define EMB    1024
#define KCB    4096      // codebook entries
#define CD     256       // code dim
#define NCODE  (B_SZ * 4)
#define EPS_BN 1e-5f

// epilogue flags
#define F_PRERELU  1
#define F_BN       2
#define F_POSTRELU 4
#define F_RES      8
#define F_SIG      16

// pack bf16(lo) into [15:0], bf16(hi) into [31:16] with ONE v_perm_b32
__device__ __forceinline__ u32 pkbf(float lo, float hi) {
  return __builtin_amdgcn_perm(__builtin_bit_cast(u32, hi),
                               __builtin_bit_cast(u32, lo), 0x07060302u);
}

// two aligned 16B chunks -> one 16xbf16 fragment (2x b128 loads)
__device__ __forceinline__ v16bf ld_frag(const u32* p0, const u32* p1) {
  v4u a = *(const v4u*)p0;
  v4u b = *(const v4u*)p1;
  return __builtin_bit_cast(v16bf,
           __builtin_shufflevector(a, b, 0, 1, 2, 3, 4, 5, 6, 7));
}

// async copy 16B global -> LDS (gfx1250; tracked with ASYNCcnt)
__device__ __forceinline__ void async_g2l_b128(u32 lds_addr, const void* gptr) {
  asm volatile("global_load_async_to_lds_b128 %0, %1, off"
               :: "v"(lds_addr), "v"((uint64_t)(uintptr_t)gptr)
               : "memory");
}
__device__ __forceinline__ void wait_async0() {
  asm volatile("s_wait_asynccnt 0x0" ::: "memory");
}

// ---------------------------------------------------------------------------
// Pre-pass: W [K][N] f32  ->  Wt [N][K] bf16 (tiled transpose + convert)
// ---------------------------------------------------------------------------
__global__ __launch_bounds__(256)
void conv_w_t(const float* __restrict__ W, u32* __restrict__ Wt, int Kd, int N)
{
  __shared__ float tile[32][33];
  const int tid = threadIdx.x;
  const int k0 = blockIdx.x * 32;
  const int n0 = blockIdx.y * 32;
  #pragma unroll
  for (int i = 0; i < 4; ++i) {
    int s  = tid + 256 * i;                  // 1024 elements
    int kk = s >> 5, nn = s & 31;
    tile[kk][nn] = W[(size_t)(k0 + kk) * N + n0 + nn];
  }
  __syncthreads();
  const int n  = tid >> 3;                   // 0..31 (output row = col of W)
  const int d2 = (tid & 7) * 2;              // dword pair base
  const int kk = d2 * 2;                     // k offset (0..28 step 4)
  v2u d;
  d.x = pkbf(tile[kk + 0][n], tile[kk + 1][n]);
  d.y = pkbf(tile[kk + 2][n], tile[kk + 3][n]);
  u32* dst = Wt + ((size_t)(n0 + n) * Kd + k0) / 2 + d2;
  *(v2u*)dst = d;
}

// ---------------------------------------------------------------------------
// Pre-pass: codebook f32 -> bf16 row-major
// ---------------------------------------------------------------------------
__global__ __launch_bounds__(256)
void conv_cb(const float* __restrict__ cb, u32* __restrict__ cbh)
{
  size_t t = (size_t)blockIdx.x * 256 + threadIdx.x;   // 4 floats / thread
  const float4 f = *(const float4*)(cb + t * 4);
  v2u d; d.x = pkbf(f.x, f.y); d.y = pkbf(f.z, f.w);
  *(v2u*)(cbh + t * 2) = d;
}

// ---------------------------------------------------------------------------
// Tiled WMMA GEMM: C[M,N] = epilogue(A[M,K] @ W[K,N]),  W given as Wt[N][K] bf16
// 256 threads = 8 waves; workgroup tile 128x128; wave tile 32x64 (2x4 WMMA)
// Double-buffered LDS pipeline with async global->LDS for B.
// ---------------------------------------------------------------------------
__global__ __launch_bounds__(256)
void gemm_wmma_bf16(const float* __restrict__ A, const u32* __restrict__ Wt,
                    const float* __restrict__ bias,
                    const float* __restrict__ g,  const float* __restrict__ bb,
                    const float* __restrict__ mn, const float* __restrict__ vr,
                    const float* __restrict__ res, float* __restrict__ C,
                    int M, int N, int Kd, int flags)
{
  __shared__ u32 lA[2][128][20];  // 128 rows x 32 bf16 (16 dw) + 4 dw pad
  __shared__ u32 lB[2][128][20];  // 128 cols x 32 bf16, K-contiguous per col

  const int tid  = threadIdx.x;
  const int wave = tid >> 5;
  const int lane = tid & 31;
  const int half = lane >> 4;
  const int lr   = lane & 15;
  const int waveM = wave >> 1;      // 0..3
  const int waveN = wave & 1;       // 0..1
  const int bm = blockIdx.y * 128;
  const int bn = blockIdx.x * 128;
  const int kdw = Kd >> 1;          // Wt row stride in dwords

  // B staging: issue async copies of k-tile k0 into buffer b
  auto stageB = [&](int k0, int b) {
    #pragma unroll
    for (int i = 0; i < 2; ++i) {
      int slot = tid + 256 * i;             // 512 x 16B chunks
      int col  = slot & 127;
      int q    = slot >> 7;                 // 0..3
      const u32* src = Wt + (size_t)(bn + col) * kdw + (k0 >> 1) + q * 4;
      async_g2l_b128((u32)(uintptr_t)&lB[b][col][q * 4], src);
    }
  };
  // A staging, split: global loads into registers ...
  auto loadA = [&](int k0, float4* fr) {
    #pragma unroll
    for (int i = 0; i < 4; ++i) {
      int slot = tid + 256 * i;             // 1024 float4 slots
      int row  = slot >> 3;
      int c4   = (slot & 7) << 2;
      const float* ap = A + (size_t)(bm + row) * Kd + k0 + c4;
      fr[i] = *(const float4*)ap;
      if (i == 0 && k0 + 64 < Kd) __builtin_prefetch(ap + 64, 0, 1);
    }
  };
  // ... then convert + store into LDS buffer b
  auto storeA = [&](const float4* fr, int b) {
    #pragma unroll
    for (int i = 0; i < 4; ++i) {
      int slot = tid + 256 * i;
      int row  = slot >> 3;
      int c4   = (slot & 7) << 2;
      float x0 = fr[i].x, x1 = fr[i].y, x2 = fr[i].z, x3 = fr[i].w;
      if (flags & F_PRERELU) {
        x0 = fmaxf(x0, 0.f); x1 = fmaxf(x1, 0.f);
        x2 = fmaxf(x2, 0.f); x3 = fmaxf(x3, 0.f);
      }
      v2u d; d.x = pkbf(x0, x1); d.y = pkbf(x2, x3);
      *(v2u*)&lA[b][row][c4 >> 1] = d;
    }
  };

  v8f acc[2][4];
  #pragma unroll
  for (int i = 0; i < 2; ++i)
    #pragma unroll
    for (int j = 0; j < 4; ++j)
      #pragma unroll
      for (int e = 0; e < 8; ++e) acc[i][j][e] = 0.0f;

  const int kTiles = Kd >> 5;

  // ---- prologue: stage tile 0 into buffer 0 ----
  stageB(0, 0);
  {
    float4 fr0[4];
    loadA(0, fr0);
    storeA(fr0, 0);
  }
  wait_async0();
  __syncthreads();

  for (int kt = 0; kt < kTiles; ++kt) {
    const int  buf  = kt & 1;
    const bool more = (kt + 1 < kTiles);

    // issue next tile's B async copies + A global loads before computing
    float4 fr[4];
    if (more) {
      stageB((kt + 1) << 5, buf ^ 1);
      loadA((kt + 1) << 5, fr);
    }

    // ---- compute on current buffer ----
    v16bf af[2];
    #pragma unroll
    for (int mf = 0; mf < 2; ++mf) {
      const int r = waveM * 32 + mf * 16 + lr;
      af[mf] = ld_frag(&lA[buf][r][half * 4], &lA[buf][r][8 + half * 4]);
    }
    #pragma unroll
    for (int nf = 0; nf < 4; ++nf) {
      const int c = waveN * 64 + nf * 16 + lr;
      v16bf bfm = ld_frag(&lB[buf][c][half * 4], &lB[buf][c][8 + half * 4]);
      #pragma unroll
      for (int mf = 0; mf < 2; ++mf)
        acc[mf][nf] = __builtin_amdgcn_wmma_f32_16x16x32_bf16(
            false, af[mf], false, bfm, (short)0, acc[mf][nf], false, false);
    }

    if (more) {
      storeA(fr, buf ^ 1);    // convert + LDS store after WMMAs issued
      wait_async0();          // B(t+1) complete (ASYNCcnt not in barrier wait)
      __syncthreads();        // one barrier per iteration (disjoint buffers)
    }
  }

  // ---- epilogue: D layout VGPR r, lane -> row = r + 8*half, col = lr ----
  #pragma unroll
  for (int nf = 0; nf < 4; ++nf) {
    const int col = bn + waveN * 64 + nf * 16 + lr;
    const float bsv = bias[col];
    float sg = 1.f, sb = 0.f;
    if (flags & F_BN) {
      const float inv = rsqrtf(vr[col] + EPS_BN);
      sg = g[col] * inv;
      sb = bb[col] - mn[col] * g[col] * inv;
    }
    #pragma unroll
    for (int mf = 0; mf < 2; ++mf) {
      const int rbase = bm + waveM * 32 + mf * 16 + half * 8;
      #pragma unroll
      for (int r = 0; r < 8; ++r) {
        const int row = rbase + r;
        float val = acc[mf][nf][r] + bsv;
        if (flags & F_BN)       val = val * sg + sb;
        if (flags & F_POSTRELU) val = fmaxf(val, 0.f);
        if (flags & F_RES)      val += res[(size_t)row * N + col];
        if (flags & F_SIG)      val = 1.0f / (1.0f + __expf(-val));
        C[(size_t)row * N + col] = val;
      }
    }
  }
}

// ---------------------------------------------------------------------------
// Codebook squared norms (f32 source)
// ---------------------------------------------------------------------------
__global__ __launch_bounds__(256)
void cb_norms(const float* __restrict__ cb, float* __restrict__ out)
{
  int n = blockIdx.x * blockDim.x + threadIdx.x;
  if (n >= KCB) return;
  const float* p = cb + (size_t)n * CD;
  float s = 0.f;
  for (int j = 0; j < CD; j += 4) {
    float4 f = *(const float4*)(p + j);
    s += f.x * f.x + f.y * f.y + f.z * f.z + f.w * f.w;
  }
  out[n] = s;
}

// ---------------------------------------------------------------------------
// VQ argmin via WMMA: per block 16 z-rows vs all 4096 codebook entries.
// codebook already bf16 -> B fragment = two global b128 loads, no converts
// ---------------------------------------------------------------------------
__global__ __launch_bounds__(256)
void vq_argmin(const float* __restrict__ z, const u32* __restrict__ cbh,
               const float* __restrict__ cbn, int* __restrict__ idx_out,
               int* __restrict__ hist)
{
  __shared__ u32   lF[16][132];    // 16 rows x 256 bf16 (128 dw) + 4 dw pad
  __shared__ float sval[8][16];
  __shared__ int   sidx[8][16];

  const int tid  = threadIdx.x;
  const int wave = tid >> 5;
  const int lane = tid & 31;
  const int half = lane >> 4;
  const int lr   = lane & 15;
  const int rowBase = blockIdx.x * 16;

  // stage 16 z rows (f32 -> packed bf16 dwords)
  #pragma unroll
  for (int i = 0; i < 4; ++i) {
    int slot = tid + 256 * i;
    int row  = slot >> 6;
    int c4   = (slot & 63) << 2;
    const float4 f = *(const float4*)(z + (size_t)(rowBase + row) * CD + c4);
    v2u d; d.x = pkbf(f.x, f.y); d.y = pkbf(f.z, f.w);
    *(v2u*)&lF[row][c4 >> 1] = d;
  }
  __syncthreads();

  // preload all 8 K-step A fragments (2x b128 each), reused for 4096 cols
  v16bf af[8];
  #pragma unroll
  for (int kt = 0; kt < 8; ++kt)
    af[kt] = ld_frag(&lF[lr][kt * 16 + half * 4],
                     &lF[lr][kt * 16 + 8 + half * 4]);

  float best[8]; int bidx[8];
  #pragma unroll
  for (int r = 0; r < 8; ++r) { best[r] = 3.4e38f; bidx[r] = 0; }

  // sweep 256 N-tiles of 16 codebook rows; 8 waves interleave
  for (int nt = wave; nt < 256; nt += 8) {
    const int n = nt * 16 + lr;                // this lane's codebook row
    v8f acc;
    #pragma unroll
    for (int e = 0; e < 8; ++e) acc[e] = 0.0f;
    #pragma unroll
    for (int kt = 0; kt < 8; ++kt) {
      const u32* cp = cbh + (size_t)n * (CD >> 1) + kt * 16 + half * 4;
      v16bf bfm = ld_frag(cp, cp + 8);         // k = 8h..8h+7, 16+8h..16+8h+7
      acc = __builtin_amdgcn_wmma_f32_16x16x32_bf16(
          false, af[kt], false, bfm, (short)0, acc, false, false);
    }
    const float cn = cbn[n];
    #pragma unroll
    for (int r = 0; r < 8; ++r) {
      float s = cn - 2.0f * acc[r];            // row = half*8 + r, col = n
      if (s < best[r]) { best[r] = s; bidx[r] = n; }
    }
  }

  // min-reduce across the 16 lanes of each half (same rows, different cols)
  #pragma unroll
  for (int m = 8; m >= 1; m >>= 1)
    #pragma unroll
    for (int r = 0; r < 8; ++r) {
      float ov = __shfl_xor(best[r], m, 32);
      int   oi = __shfl_xor(bidx[r], m, 32);
      if (ov < best[r]) { best[r] = ov; bidx[r] = oi; }
    }

  if (lr == 0) {
    #pragma unroll
    for (int r = 0; r < 8; ++r) {
      sval[wave][half * 8 + r] = best[r];
      sidx[wave][half * 8 + r] = bidx[r];
    }
  }
  __syncthreads();

  if (tid < 16) {                               // reduce across 8 waves
    float bv = sval[0][tid]; int bi = sidx[0][tid];
    #pragma unroll
    for (int w = 1; w < 8; ++w)
      if (sval[w][tid] < bv) { bv = sval[w][tid]; bi = sidx[w][tid]; }
    idx_out[rowBase + tid] = bi;
    atomicAdd(&hist[bi], 1);
  }
}

// ---------------------------------------------------------------------------
// Gather codebook rows into hq and accumulate sum((q - z)^2)
// ---------------------------------------------------------------------------
__global__ __launch_bounds__(256)
void gather_loss(const int* __restrict__ idx, const float* __restrict__ cb,
                 const float* __restrict__ z, float* __restrict__ hq,
                 float* __restrict__ lossAcc)
{
  __shared__ float red[256];
  const int b   = blockIdx.x;
  const int tid = threadIdx.x;
  float s = 0.f;
  #pragma unroll
  for (int i = 0; i < 4; ++i) {
    int j    = tid + 256 * i;
    int comb = j >> 8, jj = j & 255;
    int code = idx[b * 4 + comb];
    float q  = cb[(size_t)code * CD + jj];
    float zz = z[(size_t)b * EMB + j];
    hq[(size_t)b * EMB + j] = q;
    float d = q - zz; s += d * d;
  }
  red[tid] = s; __syncthreads();
  for (int st = 128; st > 0; st >>= 1) {
    if (tid < st) red[tid] += red[tid + st];
    __syncthreads();
  }
  if (tid == 0) atomicAdd(lossAcc, red[0]);
}

// ---------------------------------------------------------------------------
// Finalize: loss = 1.25 * mse ; perplexity = exp(-sum p log(p+1e-10))
// ---------------------------------------------------------------------------
__global__ __launch_bounds__(256)
void finalize(const int* __restrict__ hist, const float* __restrict__ lossAcc,
              float* __restrict__ out_loss, float* __restrict__ out_ppl)
{
  __shared__ float red[256];
  const int tid = threadIdx.x;
  float s = 0.f;
  for (int j = tid; j < KCB; j += 256) {
    float p = (float)hist[j] * (1.0f / (float)NCODE);
    s += p * __logf(p + 1e-10f);
  }
  red[tid] = s; __syncthreads();
  for (int st = 128; st > 0; st >>= 1) {
    if (tid < st) red[tid] += red[tid + st];
    __syncthreads();
  }
  if (tid == 0) {
    *out_ppl  = __expf(-red[0]);
    *out_loss = 1.25f * (*lossAcc) / (float)((size_t)B_SZ * EMB);
  }
}

// ---------------------------------------------------------------------------
extern "C" void kernel_launch(void* const* d_in, const int* in_sizes, int n_in,
                              void* d_out, int out_size, void* d_ws, size_t ws_size,
                              hipStream_t stream)
{
  const float* x    = (const float*)d_in[0];
  const float* eW1  = (const float*)d_in[1];
  const float* eb1  = (const float*)d_in[2];
  const float* bn1g = (const float*)d_in[3];
  const float* bn1b = (const float*)d_in[4];
  const float* bn1m = (const float*)d_in[5];
  const float* bn1v = (const float*)d_in[6];
  const float* eW2  = (const float*)d_in[7];
  const float* eb2  = (const float*)d_in[8];
  const float* rbW  = (const float*)d_in[9];
  const float* rbB  = (const float*)d_in[10];
  const float* rbG  = (const float*)d_in[11];
  const float* rbBe = (const float*)d_in[12];
  const float* rbM  = (const float*)d_in[13];
  const float* rbV  = (const float*)d_in[14];
  const float* cb   = (const float*)d_in[15];
  const float* dW   = (const float*)d_in[16];
  const float* db   = (const float*)d_in[17];

  float* out      = (float*)d_out;
  float* out_loss = out;
  float* x_recon  = out + 1;
  float* out_ppl  = out + 1 + (size_t)B_SZ * IN_SZ;

  // workspace layout
  char*  ws   = (char*)d_ws;
  size_t off  = 0;
  float* actA = (float*)(ws + off); off += (size_t)B_SZ * EMB * 4;     // 16 MB
  float* actB = (float*)(ws + off); off += (size_t)B_SZ * EMB * 4;     // 16 MB
  u32* wtE1   = (u32*)(ws + off);   off += (size_t)IN_SZ * EMB * 2;    // 16 MB
  u32* wtE2   = (u32*)(ws + off);   off += (size_t)EMB * EMB * 2;      //  2 MB
  u32* wtRB   = (u32*)(ws + off);   off += (size_t)5 * EMB * EMB * 2;  // 10 MB
  u32* wtDec  = (u32*)(ws + off);   off += (size_t)EMB * IN_SZ * 2;    // 16 MB
  u32* cbh    = (u32*)(ws + off);   off += (size_t)KCB * CD * 2;       //  2 MB
  int*   idx     = (int*)(ws + off);   off += NCODE * 4;
  float* cbn     = (float*)(ws + off); off += KCB * 4;
  int*   hist    = (int*)(ws + off);   off += KCB * 4;
  float* lossAcc = (float*)(ws + off); off += 64;

  hipMemsetAsync(hist, 0, KCB * sizeof(int), stream);
  hipMemsetAsync(lossAcc, 0, sizeof(float), stream);

  const dim3 blk(256);
  const dim3 gEmb(EMB / 128, B_SZ / 128);
  const dim3 gIn(IN_SZ / 128, B_SZ / 128);
  const size_t rbWst = (size_t)EMB * EMB / 2;  // rb weight stride in dwords

  // ---- weight pre-conversion (f32 -> bf16, transposed to [N][K]) ----
  conv_w_t<<<dim3(IN_SZ / 32, EMB / 32), blk, 0, stream>>>(eW1, wtE1, IN_SZ, EMB);
  conv_w_t<<<dim3(EMB / 32, EMB / 32), blk, 0, stream>>>(eW2, wtE2, EMB, EMB);
  for (int i = 0; i < 5; ++i)
    conv_w_t<<<dim3(EMB / 32, EMB / 32), blk, 0, stream>>>(
        rbW + (size_t)i * EMB * EMB, wtRB + i * rbWst, EMB, EMB);
  conv_w_t<<<dim3(EMB / 32, IN_SZ / 32), blk, 0, stream>>>(dW, wtDec, EMB, IN_SZ);
  conv_cb<<<dim3(KCB * CD / 1024), blk, 0, stream>>>(cb, cbh);
  cb_norms<<<dim3(KCB / 256), blk, 0, stream>>>(cb, cbn);

  // ---- encoder ----
  gemm_wmma_bf16<<<gEmb, blk, 0, stream>>>(x, wtE1, eb1, bn1g, bn1b, bn1m, bn1v,
                                           nullptr, actA, B_SZ, EMB, IN_SZ,
                                           F_BN | F_POSTRELU);
  gemm_wmma_bf16<<<gEmb, blk, 0, stream>>>(actA, wtE2, eb2, nullptr, nullptr,
                                           nullptr, nullptr, nullptr, actB,
                                           B_SZ, EMB, EMB, 0);
  gemm_wmma_bf16<<<gEmb, blk, 0, stream>>>(actB, wtRB + 0 * rbWst, rbB + 0 * EMB,
                                           rbG + 0 * EMB, rbBe + 0 * EMB,
                                           rbM + 0 * EMB, rbV + 0 * EMB,
                                           actB, actA, B_SZ, EMB, EMB,
                                           F_PRERELU | F_BN | F_RES);
  gemm_wmma_bf16<<<gEmb, blk, 0, stream>>>(actA, wtRB + 1 * rbWst, rbB + 1 * EMB,
                                           rbG + 1 * EMB, rbBe + 1 * EMB,
                                           rbM + 1 * EMB, rbV + 1 * EMB,
                                           actA, actB, B_SZ, EMB, EMB,
                                           F_PRERELU | F_BN | F_RES);
  // z lives in actB

  // ---- vector quantizer ----
  vq_argmin<<<dim3(NCODE / 16), blk, 0, stream>>>(actB, cbh, cbn, idx, hist);
  gather_loss<<<dim3(B_SZ), blk, 0, stream>>>(idx, cb, actB, actA, lossAcc);
  // hq lives in actA

  // ---- decoder ----
  gemm_wmma_bf16<<<gEmb, blk, 0, stream>>>(actA, wtRB + 2 * rbWst, rbB + 2 * EMB,
                                           rbG + 2 * EMB, rbBe + 2 * EMB,
                                           rbM + 2 * EMB, rbV + 2 * EMB,
                                           actA, actB, B_SZ, EMB, EMB,
                                           F_PRERELU | F_BN | F_RES);
  gemm_wmma_bf16<<<gEmb, blk, 0, stream>>>(actB, wtRB + 3 * rbWst, rbB + 3 * EMB,
                                           rbG + 3 * EMB, rbBe + 3 * EMB,
                                           rbM + 3 * EMB, rbV + 3 * EMB,
                                           actB, actA, B_SZ, EMB, EMB,
                                           F_PRERELU | F_BN | F_RES);
  gemm_wmma_bf16<<<gEmb, blk, 0, stream>>>(actA, wtRB + 4 * rbWst, rbB + 4 * EMB,
                                           rbG + 4 * EMB, rbBe + 4 * EMB,
                                           rbM + 4 * EMB, rbV + 4 * EMB,
                                           actA, actB, B_SZ, EMB, EMB,
                                           F_PRERELU | F_BN | F_RES);
  gemm_wmma_bf16<<<gIn, blk, 0, stream>>>(actB, wtDec, db, nullptr, nullptr,
                                          nullptr, nullptr, nullptr, x_recon,
                                          B_SZ, IN_SZ, EMB, F_SIG);

  // ---- scalars ----
  finalize<<<dim3(1), blk, 0, stream>>>(hist, lossAcc, out_loss, out_ppl);
}
